// GINNet_63350767616006
// MI455X (gfx1250) — matrix operand order repaired and checked by reading, not compiled, not run on hardware
//
#include <hip/hip_runtime.h>

#define N_NODES    100000
#define N_EDGES    3200000
#define IN_DIM     1024
#define HID        128
#define N_LAYERS   3
#define NUM_GRAPHS 512
#define LN_EPS     1e-5f
#define NB_SCAN    ((N_NODES + 255) / 256)   // 391 scan blocks

typedef __attribute__((ext_vector_type(16))) __bf16 bf16x16;
typedef __attribute__((ext_vector_type(8)))  __bf16 bf16x8;
typedef __attribute__((ext_vector_type(4)))  __bf16 bf16x4;
typedef __attribute__((ext_vector_type(8)))  float  f32x8;

__device__ __forceinline__ __bf16 f2bf(float f) {
  union { float f; unsigned u; } v; v.f = f;
  unsigned r = v.u + 0x7FFFu + ((v.u >> 16) & 1u);   // round-to-nearest-even
  unsigned short s = (unsigned short)(r >> 16);
  return __builtin_bit_cast(__bf16, s);
}

__device__ __forceinline__ float gelu_exact(float x) {
  return 0.5f * x * (1.0f + erff(x * 0.70710678118654752f));
}

// ---------------------------------------------------------------------------
// Pack W [l][k][n] (f32) -> Wp [l][n][k] (bf16): matches WMMA B-fragment reads
// ---------------------------------------------------------------------------
__global__ void pack_transpose_kernel(const float* __restrict__ W,
                                      __bf16* __restrict__ Wp,
                                      int K, int N, int total) {
  int t = blockIdx.x * 256 + threadIdx.x;
  if (t >= total) return;
  int kn = K * N;
  int l = t / kn;
  int rem = t - l * kn;
  int k = rem / N;
  int n = rem - k * N;
  Wp[(size_t)l * kn + (size_t)n * K + k] = f2bf(W[t]);
}

// ---------------------------------------------------------------------------
// CSR build: degree count -> two-level exclusive scan -> scatter by dst
// ---------------------------------------------------------------------------
__global__ void zero_u32_kernel(unsigned* p, int n) {
  int t = blockIdx.x * 256 + threadIdx.x;
  if (t < n) p[t] = 0u;
}

__global__ void deg_count_kernel(const int* __restrict__ ei, int* __restrict__ deg,
                                 int nEdges) {
  int e = blockIdx.x * 256 + threadIdx.x;
  if (e < nEdges) atomicAdd(&deg[ei[nEdges + e]], 1);
}

__global__ void block_sum_kernel(const int* __restrict__ deg, int* __restrict__ bsum,
                                 int n) {
  int i = blockIdx.x * 256 + threadIdx.x;
  int v = (i < n) ? deg[i] : 0;
  #pragma unroll
  for (int m = 16; m >= 1; m >>= 1) v += __shfl_xor(v, m, 32);
  __shared__ int ws[8];
  if ((threadIdx.x & 31) == 0) ws[threadIdx.x >> 5] = v;
  __syncthreads();
  if (threadIdx.x == 0) {
    int s = 0;
    for (int w = 0; w < 8; ++w) s += ws[w];
    bsum[blockIdx.x] = s;
  }
}

// single block, 512 threads: in-place exclusive scan of nb (<512) block sums
__global__ __launch_bounds__(512) void scan_small_kernel(int* bsum, int nb) {
  __shared__ int tmp[512];
  int t = threadIdx.x;
  int v = (t < nb) ? bsum[t] : 0;
  tmp[t] = v;
  __syncthreads();
  for (int d = 1; d < 512; d <<= 1) {
    int add = (t >= d) ? tmp[t - d] : 0;
    __syncthreads();
    tmp[t] += add;
    __syncthreads();
  }
  if (t < nb) bsum[t] = tmp[t] - v;   // exclusive
}

__global__ void scan_block_kernel(const int* __restrict__ deg, const int* __restrict__ boff,
                                  int* __restrict__ off, int n, int nEdges) {
  int i = blockIdx.x * 256 + threadIdx.x;
  int v = (i < n) ? deg[i] : 0;
  int lane = threadIdx.x & 31, wv = threadIdx.x >> 5;
  int inc = v;
  #pragma unroll
  for (int d = 1; d < 32; d <<= 1) {
    int t = __shfl_up(inc, d, 32);
    if (lane >= d) inc += t;
  }
  __shared__ int ws[8];
  if (lane == 31) ws[wv] = inc;
  __syncthreads();
  int wadd = 0;
  for (int w = 0; w < 8; ++w)
    if (w < wv) wadd += ws[w];
  if (i < n) off[i] = inc - v + wadd + boff[blockIdx.x];
  if (i == 0) off[n] = nEdges;
}

__global__ void copy_i32_kernel(const int* __restrict__ a, int* __restrict__ b, int n) {
  int t = blockIdx.x * 256 + threadIdx.x;
  if (t < n) b[t] = a[t];
}

__global__ void scatter_kernel(const int* __restrict__ ei, int* __restrict__ cursor,
                               int* __restrict__ nbr, int nEdges) {
  int e = blockIdx.x * 256 + threadIdx.x;
  if (e >= nEdges) return;
  int dst = ei[nEdges + e];
  int p = atomicAdd(&cursor[dst], 1);
  nbr[p] = ei[e];                       // cursor pre-initialized to row offsets
}

// ---------------------------------------------------------------------------
// Featurizer: x = gelu(nf @ W_feat + b). 16-row tile, 8 waves = 8 n-tiles,
// split-K dual accumulators for back-to-back independent WMMA chains.
// ---------------------------------------------------------------------------
#define A_STR (IN_DIM + 8)   // halves; 2064B row stride -> conflict-free b128 LDS reads

__global__ __launch_bounds__(256) void featurize_kernel(
    const float* __restrict__ nf, const __bf16* __restrict__ Wp,
    const float* __restrict__ bias, float* __restrict__ x) {
  __shared__ __align__(16) __bf16 As[16 * A_STR];
  const int tid = threadIdx.x;
  const int m0  = blockIdx.x * 16;

  const float4* nf4 = (const float4*)(nf + (size_t)m0 * IN_DIM);
  #pragma unroll
  for (int i = 0; i < 16; ++i) {
    int idx = tid + i * 256;          // 4096 float4 slots
    int r   = idx >> 8;               // 256 float4 per row
    int c4  = idx & 255;
    float4 v = nf4[r * 256 + c4];
    bf16x4 bv; bv[0]=f2bf(v.x); bv[1]=f2bf(v.y); bv[2]=f2bf(v.z); bv[3]=f2bf(v.w);
    *(bf16x4*)&As[r * A_STR + c4 * 4] = bv;
  }
  __syncthreads();

  const int lane = tid & 31;
  const int wv   = tid >> 5;
  const int l16  = lane & 15;
  const int hi   = lane >> 4;
  const int nb   = wv * 16;
  const __bf16* __restrict__ Brow = Wp + (size_t)(nb + l16) * IN_DIM + hi * 16;
  const __bf16* __restrict__ Ar   = As + l16 * A_STR + hi * 16;

  f32x8 acc0 = {}, acc1 = {};
  #pragma unroll 2
  for (int k0 = 0; k0 < IN_DIM; k0 += 64) {       // two independent K-chains
    __builtin_prefetch(Brow + k0 + 256, 0, 3);    // global_prefetch_b8 on weights
    bf16x16 a0, b0, a1, b1;
    {
      bf16x8 alo = *(const bf16x8*)(Ar + k0);
      bf16x8 ahi = *(const bf16x8*)(Ar + k0 + 8);
      bf16x8 blo = *(const bf16x8*)(Brow + k0);
      bf16x8 bhi = *(const bf16x8*)(Brow + k0 + 8);
      #pragma unroll
      for (int i = 0; i < 8; ++i) { a0[i]=alo[i]; a0[i+8]=ahi[i]; b0[i]=blo[i]; b0[i+8]=bhi[i]; }
    }
    {
      bf16x8 alo = *(const bf16x8*)(Ar + k0 + 32);
      bf16x8 ahi = *(const bf16x8*)(Ar + k0 + 40);
      bf16x8 blo = *(const bf16x8*)(Brow + k0 + 32);
      bf16x8 bhi = *(const bf16x8*)(Brow + k0 + 40);
      #pragma unroll
      for (int i = 0; i < 8; ++i) { a1[i]=alo[i]; a1[i+8]=ahi[i]; b1[i]=blo[i]; b1[i+8]=bhi[i]; }
    }
    acc0 = __builtin_amdgcn_wmma_f32_16x16x32_bf16(false, a0, false, b0, (short)0, acc0,
                                                   false, false);
    acc1 = __builtin_amdgcn_wmma_f32_16x16x32_bf16(false, a1, false, b1, (short)0, acc1,
                                                   false, false);
  }

  const int col = nb + l16;
  const float bn = bias[col];
  #pragma unroll
  for (int j = 0; j < 8; ++j) {
    int row = m0 + j + hi * 8;
    x[(size_t)row * HID + col] = gelu_exact(acc0[j] + acc1[j] + bn);
  }
}

// ---------------------------------------------------------------------------
// GIN aggregation via CSR gather (no atomics): h[n] = x[n] + sum_{s in N(n)} x[s]
// One wave per node; lane holds a float4 column slice.
// ---------------------------------------------------------------------------
__global__ __launch_bounds__(256) void gather_aggr_kernel(
    const float* __restrict__ x, const int* __restrict__ off,
    const int* __restrict__ nbr, float* __restrict__ h) {
  int node = blockIdx.x * 8 + (threadIdx.x >> 5);
  if (node >= N_NODES) return;
  int lane = threadIdx.x & 31;
  int c = lane * 4;
  int beg = off[node], end = off[node + 1];
  float4 acc = *(const float4*)&x[(size_t)node * HID + c];   // self term (eps=0)
  for (int p = beg; p < end; ++p) {
    int s = nbr[p];
    const float4 v = *(const float4*)&x[(size_t)s * HID + c];
    acc.x += v.x; acc.y += v.y; acc.z += v.z; acc.w += v.w;
  }
  *(float4*)&h[(size_t)node * HID + c] = acc;
}

// ---------------------------------------------------------------------------
// MLP step: out = gelu(LN(in @ W + b)) [+ res], split-K dual WMMA accumulators
// ---------------------------------------------------------------------------
#define AH_STR (HID + 8)
#define CS_STR (HID + 4)

__global__ __launch_bounds__(256) void mlp_ln_gelu_kernel(
    const float* __restrict__ in, const __bf16* __restrict__ Wp,
    const float* __restrict__ bias, const float* __restrict__ gam,
    const float* __restrict__ bet, const float* __restrict__ res,
    float* __restrict__ outA) {
  __shared__ __align__(16) __bf16 As[16 * AH_STR];
  __shared__ float Cs[16 * CS_STR];
  const int tid = threadIdx.x;
  const int m0  = blockIdx.x * 16;

  const float4* in4 = (const float4*)(in + (size_t)m0 * HID);
  #pragma unroll
  for (int i = 0; i < 2; ++i) {
    int idx = tid + i * 256;          // 512 float4 slots
    int r   = idx >> 5;               // 32 float4 per row
    int c4  = idx & 31;
    float4 v = in4[r * 32 + c4];
    bf16x4 bv; bv[0]=f2bf(v.x); bv[1]=f2bf(v.y); bv[2]=f2bf(v.z); bv[3]=f2bf(v.w);
    *(bf16x4*)&As[r * AH_STR + c4 * 4] = bv;
  }
  __syncthreads();

  const int lane = tid & 31;
  const int wv   = tid >> 5;
  const int l16  = lane & 15;
  const int hi   = lane >> 4;
  const int nb   = wv * 16;
  const __bf16* Brow = Wp + (nb + l16) * HID + hi * 16;
  const __bf16* Ar   = As + l16 * AH_STR + hi * 16;

  f32x8 acc0 = {}, acc1 = {};
  #pragma unroll
  for (int k0 = 0; k0 < HID; k0 += 64) {          // two independent K-chains
    bf16x16 a0, b0, a1, b1;
    {
      bf16x8 alo = *(const bf16x8*)(Ar + k0);
      bf16x8 ahi = *(const bf16x8*)(Ar + k0 + 8);
      bf16x8 blo = *(const bf16x8*)(Brow + k0);
      bf16x8 bhi = *(const bf16x8*)(Brow + k0 + 8);
      #pragma unroll
      for (int i = 0; i < 8; ++i) { a0[i]=alo[i]; a0[i+8]=ahi[i]; b0[i]=blo[i]; b0[i+8]=bhi[i]; }
    }
    {
      bf16x8 alo = *(const bf16x8*)(Ar + k0 + 32);
      bf16x8 ahi = *(const bf16x8*)(Ar + k0 + 40);
      bf16x8 blo = *(const bf16x8*)(Brow + k0 + 32);
      bf16x8 bhi = *(const bf16x8*)(Brow + k0 + 40);
      #pragma unroll
      for (int i = 0; i < 8; ++i) { a1[i]=alo[i]; a1[i+8]=ahi[i]; b1[i]=blo[i]; b1[i+8]=bhi[i]; }
    }
    acc0 = __builtin_amdgcn_wmma_f32_16x16x32_bf16(false, a0, false, b0, (short)0, acc0,
                                                   false, false);
    acc1 = __builtin_amdgcn_wmma_f32_16x16x32_bf16(false, a1, false, b1, (short)0, acc1,
                                                   false, false);
  }

  // Exchange C through LDS for row-wise LayerNorm
  const int col = nb + l16;
  const float bn = bias[col];
  #pragma unroll
  for (int j = 0; j < 8; ++j)
    Cs[(j + hi * 8) * CS_STR + col] = acc0[j] + acc1[j] + bn;
  __syncthreads();

  const int r  = tid >> 4;            // row 0..15 (16 threads per row)
  const int cb = (tid & 15) * 8;
  float vals[8];
  float s = 0.f, s2 = 0.f;
  #pragma unroll
  for (int i = 0; i < 8; ++i) {
    float v = Cs[r * CS_STR + cb + i];
    vals[i] = v; s += v; s2 += v * v;
  }
  #pragma unroll
  for (int m = 1; m < 16; m <<= 1) {  // reduce within 16-lane half-groups (wave32)
    s  += __shfl_xor(s,  m, 32);
    s2 += __shfl_xor(s2, m, 32);
  }
  const float mean = s * (1.0f / HID);
  const float var  = s2 * (1.0f / HID) - mean * mean;
  const float rstd = rsqrtf(var + LN_EPS);

  const size_t rowo = (size_t)(m0 + r) * HID + cb;
  #pragma unroll
  for (int i = 0; i < 8; ++i) {
    float v = (vals[i] - mean) * rstd * gam[cb + i] + bet[cb + i];
    v = gelu_exact(v);
    if (res) v += res[rowo + i];
    outA[rowo + i] = v;
  }
}

// ---------------------------------------------------------------------------
// Pool: batch is sorted -> per-graph binary search + deterministic reduction
// ---------------------------------------------------------------------------
__device__ __forceinline__ int lower_bound_i(const int* a, int n, int key) {
  int lo = 0, hi = n;
  while (lo < hi) {
    int mid = (lo + hi) >> 1;
    if (a[mid] < key) lo = mid + 1; else hi = mid;
  }
  return lo;
}

__global__ __launch_bounds__(128) void pool_kernel(const float* __restrict__ x,
                                                   const int* __restrict__ batch,
                                                   float* __restrict__ out) {
  int g = blockIdx.x;
  int lo = lower_bound_i(batch, N_NODES, g);
  int hi = lower_bound_i(batch, N_NODES, g + 1);
  int c = threadIdx.x;
  float acc = 0.f;
  for (int n = lo; n < hi; ++n) acc += x[(size_t)n * HID + c];
  out[(size_t)g * HID + c] = acc;
}

// ---------------------------------------------------------------------------
extern "C" void kernel_launch(void* const* d_in, const int* in_sizes, int n_in,
                              void* d_out, int out_size, void* d_ws, size_t ws_size,
                              hipStream_t stream) {
  (void)in_sizes; (void)n_in; (void)out_size; (void)ws_size;
  const float* nf    = (const float*)d_in[0];
  const int*   ei    = (const int*)d_in[1];
  const int*   batch = (const int*)d_in[2];
  const float* Wf    = (const float*)d_in[3];
  const float* bfeat = (const float*)d_in[4];
  const float* Ws1   = (const float*)d_in[5];
  const float* bs1   = (const float*)d_in[6];
  const float* g1    = (const float*)d_in[7];
  const float* be1   = (const float*)d_in[8];
  const float* Ws2   = (const float*)d_in[9];
  const float* bs2   = (const float*)d_in[10];
  const float* g2    = (const float*)d_in[11];
  const float* be2   = (const float*)d_in[12];
  float* out = (float*)d_out;

  char* ws = (char*)d_ws;
  size_t cur = 0;
  auto alloc = [&](size_t bytes) {
    size_t p = cur;
    cur = (cur + bytes + 255) & ~(size_t)255;
    return p;
  };
  float*  x    = (float*)(ws + alloc((size_t)N_NODES * HID * sizeof(float)));
  float*  h    = (float*)(ws + alloc((size_t)N_NODES * HID * sizeof(float)));
  __bf16* Wfp  = (__bf16*)(ws + alloc((size_t)IN_DIM * HID * 2));
  __bf16* W1p  = (__bf16*)(ws + alloc((size_t)N_LAYERS * HID * HID * 2));
  __bf16* W2p  = (__bf16*)(ws + alloc((size_t)N_LAYERS * HID * HID * 2));
  int*    deg  = (int*)(ws + alloc((size_t)N_NODES * 4));
  int*    off  = (int*)(ws + alloc((size_t)(N_NODES + 1) * 4));
  int*    curs = (int*)(ws + alloc((size_t)N_NODES * 4));
  int*    bsum = (int*)(ws + alloc((size_t)NB_SCAN * 4));
  int*    nbr  = (int*)(ws + alloc((size_t)N_EDGES * 4));

  // Weight packing (bf16, transposed)
  pack_transpose_kernel<<<(IN_DIM * HID + 255) / 256, 256, 0, stream>>>(
      Wf, Wfp, IN_DIM, HID, IN_DIM * HID);
  pack_transpose_kernel<<<(N_LAYERS * HID * HID + 255) / 256, 256, 0, stream>>>(
      Ws1, W1p, HID, HID, N_LAYERS * HID * HID);
  pack_transpose_kernel<<<(N_LAYERS * HID * HID + 255) / 256, 256, 0, stream>>>(
      Ws2, W2p, HID, HID, N_LAYERS * HID * HID);

  // CSR build (by destination), reused for all 3 layers
  zero_u32_kernel<<<(N_NODES + 255) / 256, 256, 0, stream>>>((unsigned*)deg, N_NODES);
  deg_count_kernel<<<(N_EDGES + 255) / 256, 256, 0, stream>>>(ei, deg, N_EDGES);
  block_sum_kernel<<<NB_SCAN, 256, 0, stream>>>(deg, bsum, N_NODES);
  scan_small_kernel<<<1, 512, 0, stream>>>(bsum, NB_SCAN);
  scan_block_kernel<<<NB_SCAN, 256, 0, stream>>>(deg, bsum, off, N_NODES, N_EDGES);
  copy_i32_kernel<<<(N_NODES + 255) / 256, 256, 0, stream>>>(off, curs, N_NODES);
  scatter_kernel<<<(N_EDGES + 255) / 256, 256, 0, stream>>>(ei, curs, nbr, N_EDGES);

  // x = gelu(nf @ W_feat + b)
  featurize_kernel<<<N_NODES / 16, 256, 0, stream>>>(nf, Wfp, bfeat, x);

  for (int l = 0; l < N_LAYERS; ++l) {
    // h = x + sum_{neighbors} x  (pure gather, no atomics)
    gather_aggr_kernel<<<(N_NODES + 7) / 8, 256, 0, stream>>>(x, off, nbr, h);
    // h <- gelu(LN(h @ W1 + b1))  (in-place: block reads its rows before writing)
    mlp_ln_gelu_kernel<<<N_NODES / 16, 256, 0, stream>>>(
        h, W1p + (size_t)l * HID * HID, bs1 + l * HID, g1 + l * HID, be1 + l * HID,
        nullptr, h);
    // x <- gelu(LN(h @ W2 + b2)) + x
    mlp_ln_gelu_kernel<<<N_NODES / 16, 256, 0, stream>>>(
        h, W2p + (size_t)l * HID * HID, bs2 + l * HID, g2 + l * HID, be2 + l * HID,
        x, x);
  }

  // Deterministic global_add_pool over sorted batch
  pool_kernel<<<NUM_GRAPHS, 128, 0, stream>>>(x, batch, out);
}
